// PrototypeAttentionLoss_52750788329576
// MI455X (gfx1250) — matrix operand order repaired and checked by reading, not compiled, not run on hardware
//
#include <hip/hip_runtime.h>

// ---------------------------------------------------------------------------
// PrototypeAttentionLoss for MI455X (gfx1250, wave32, WMMA bf16 16x16x32).
// Algebraic folding removes the K/V projection GEMMs (550 GF -> ~37 GF),
// bringing the kernel to the HBM roofline (~512MB band reads @ 23.3 TB/s).
// ---------------------------------------------------------------------------

typedef __attribute__((ext_vector_type(16))) __bf16 bf16x16;
typedef __attribute__((ext_vector_type(8)))  __bf16 bf16x8;
typedef __attribute__((ext_vector_type(8)))  float  f32x8;

#define DD   512
#define BN   64
#define BB   2048
#define CN   8
#define K2D  1024           // concat K (real|imag)

// ws layout (bytes), all 256B aligned
constexpr size_t OFF_QR     = 0;          // 8*512*4
constexpr size_t OFF_QI     = 16384;      // 8*512*4
constexpr size_t OFF_S0     = 32768;      // 16*4 (padded)
constexpr size_t OFF_WSCORE = 33024;      // 16*1024*2 bf16
constexpr size_t OFF_WCATR  = 65792;      // 512*1024*2 bf16
constexpr size_t OFF_WCATI  = 1114368;    // 512*1024*2 bf16
constexpr size_t OFF_UCAT   = 2162944;    // 16384*1024*2 bf16
constexpr size_t OFF_LOSS   = 35717376;   // 2048*4

__device__ inline unsigned short f2bfu(float f) {
  unsigned u = __builtin_bit_cast(unsigned, f);
  u += 0x7FFFu + ((u >> 16) & 1u);        // round-to-nearest-even
  return (unsigned short)(u >> 16);
}
__device__ inline __bf16 u2bf(unsigned short s) { return __builtin_bit_cast(__bf16, s); }

// --------------------------- k1a: q = cv_linear(proto) ----------------------
__global__ void __launch_bounds__(256)
k1a_q(const float* __restrict__ proto,
      const float* __restrict__ Wqr, const float* __restrict__ Wqi,
      const float* __restrict__ bqr, const float* __restrict__ bqi,
      float* __restrict__ qr, float* __restrict__ qi)
{
  const int c = blockIdx.x;
  const float* Pr = proto + (size_t)c * DD;
  const float* Pi = proto + (size_t)CN * DD + (size_t)c * DD;
  for (int o = threadIdx.x; o < DD; o += blockDim.x) {
    float ar = bqr[o], ai = bqi[o];
    const float* wr = Wqr + (size_t)o * DD;
    const float* wi = Wqi + (size_t)o * DD;
    for (int k = 0; k < DD; ++k) {
      float xr = Pr[k], xi = Pi[k];
      ar += xr * wr[k] - xi * wi[k];
      ai += xr * wi[k] + xi * wr[k];
    }
    qr[(size_t)c * DD + o] = ar;
    qi[(size_t)c * DD + o] = ai;
  }
}

// ------------------- k1b: fold Wk through q -> Wscore (bf16) ----------------
__global__ void __launch_bounds__(256)
k1b_fold(const float* __restrict__ qr, const float* __restrict__ qi,
         const float* __restrict__ Wkr, const float* __restrict__ Wki,
         const float* __restrict__ bkr, const float* __restrict__ bki,
         unsigned short* __restrict__ wscore, float* __restrict__ s0)
{
  const int c = blockIdx.x;
  if (c >= CN) {                          // zero-pad columns 8..15 of N tile
    for (int k = threadIdx.x; k < K2D; k += blockDim.x)
      wscore[(size_t)c * K2D + k] = 0;
    if (threadIdx.x == 0) s0[c] = 0.0f;
    return;
  }
  __shared__ float qsr[DD], qsi[DD];
  for (int o = threadIdx.x; o < DD; o += blockDim.x) {
    qsr[o] = qr[(size_t)c * DD + o];
    qsi[o] = qi[(size_t)c * DD + o];
  }
  __syncthreads();
  for (int i = threadIdx.x; i < DD; i += blockDim.x) {
    float aA = 0.0f, aB = 0.0f;
    for (int o = 0; o < DD; ++o) {
      float wr = Wkr[(size_t)o * DD + i], wi = Wki[(size_t)o * DD + i];
      aA += qsr[o] * wr + qsi[o] * wi;    // coeff of band_r
      aB += qsi[o] * wr - qsr[o] * wi;    // coeff of band_i
    }
    wscore[(size_t)c * K2D + i]      = f2bfu(aA);
    wscore[(size_t)c * K2D + DD + i] = f2bfu(aB);
  }
  if (threadIdx.x == 0) {
    float s = 0.0f;
    for (int o = 0; o < DD; ++o) s += qsr[o] * bkr[o] + qsi[o] * bki[o];
    s0[c] = s;
  }
}

// -------------------- k1c: pack Wv -> WcatR/WcatI (bf16) --------------------
__global__ void __launch_bounds__(256)
k1c_packWv(const float* __restrict__ Wvr, const float* __restrict__ Wvi,
           unsigned short* __restrict__ wcatR, unsigned short* __restrict__ wcatI)
{
  const int o = blockIdx.x;
  const float* wr = Wvr + (size_t)o * DD;
  const float* wi = Wvi + (size_t)o * DD;
  for (int k = threadIdx.x; k < K2D; k += blockDim.x) {
    float vr, vi;
    if (k < DD) { vr =  wr[k];       vi = wi[k]; }
    else        { vr = -wi[k - DD];  vi = wr[k - DD]; }
    wcatR[(size_t)o * K2D + k] = f2bfu(vr);
    wcatI[(size_t)o * K2D + k] = f2bfu(vi);
  }
}

// ---- k2: per-batch scores (WMMA) -> softmax -> attn out + U accumulation ---
__global__ void __launch_bounds__(256)
k2_attn_u(const float* __restrict__ band_r, const float* __restrict__ band_i,
          const unsigned short* __restrict__ wscore, const float* __restrict__ s0,
          float* __restrict__ attn_out, unsigned short* __restrict__ ucat)
{
  const int b    = blockIdx.x;
  const int tid  = threadIdx.x;
  const int wave = tid >> 5;
  const int lane = tid & 31;
  const int half = lane >> 4;
  const int lcol = lane & 15;

  __shared__ float sc[BN][16];            // scores (n, class)
  __shared__ float aw[CN][BN];            // attention weights

  const float* br = band_r + (size_t)b * BN * DD;
  const float* bi = band_i + (size_t)b * BN * DD;

  // ---- scores: 4 waves, each one 16-row n-tile, K = 1024 concat(r,i) ----
  if (wave < 4) {
    const int n0 = wave * 16;
    f32x8 acc = {};
    const float* rowr = br + (size_t)(n0 + lcol) * DD;
    const float* rowi = bi + (size_t)(n0 + lcol) * DD;
    for (int kb = 0; kb < K2D; kb += 32) {
      const int k0 = kb + half * 16;      // A-frag low chunk  (K 0-7 / 16-23)
      const int k1 = kb + half * 16 + 8;  // A-frag high chunk (K 8-15 / 24-31)
      const float* p0 = (k0 < DD) ? (rowr + k0) : (rowi + k0 - DD);
      const float* p1 = (k1 < DD) ? (rowr + k1) : (rowi + k1 - DD);
      float4 f0a = *(const float4*)(p0);
      float4 f0b = *(const float4*)(p0 + 4);
      float4 f1a = *(const float4*)(p1);
      float4 f1b = *(const float4*)(p1 + 4);
      bf16x16 af;
      af[0]  = u2bf(f2bfu(f0a.x)); af[1]  = u2bf(f2bfu(f0a.y));
      af[2]  = u2bf(f2bfu(f0a.z)); af[3]  = u2bf(f2bfu(f0a.w));
      af[4]  = u2bf(f2bfu(f0b.x)); af[5]  = u2bf(f2bfu(f0b.y));
      af[6]  = u2bf(f2bfu(f0b.z)); af[7]  = u2bf(f2bfu(f0b.w));
      af[8]  = u2bf(f2bfu(f1a.x)); af[9]  = u2bf(f2bfu(f1a.y));
      af[10] = u2bf(f2bfu(f1a.z)); af[11] = u2bf(f2bfu(f1a.w));
      af[12] = u2bf(f2bfu(f1b.x)); af[13] = u2bf(f2bfu(f1b.y));
      af[14] = u2bf(f2bfu(f1b.z)); af[15] = u2bf(f2bfu(f1b.w));
      bf16x16 bfm = *(const bf16x16*)(wscore + (size_t)lcol * K2D + kb + half * 16);
      acc = __builtin_amdgcn_wmma_f32_16x16x32_bf16(false, af, false, bfm,
                                                    (short)0, acc, false, false);
    }
    const float sb = s0[lcol];
#pragma unroll
    for (int r = 0; r < 8; ++r) {
      int row = n0 + r + 8 * half;
      sc[row][lcol] = 0.125f * (acc[r] + sb);   // SCALE = 64^-0.5
    }
  }
  __syncthreads();

  // ---- softmax over bands (axis n), one thread per class ----
  if (tid < CN) {
    const int c = tid;
    float m = -1e30f;
    for (int n = 0; n < BN; ++n) m = fmaxf(m, sc[n][c]);
    float s = 0.0f;
    for (int n = 0; n < BN; ++n) { float e = __expf(sc[n][c] - m); sc[n][c] = e; s += e; }
    float inv = 1.0f / s;
    for (int n = 0; n < BN; ++n) aw[c][n] = sc[n][c] * inv;
  }
  __syncthreads();

  // ---- write attn (coalesced) ----
  const float* awf = &aw[0][0];
  for (int j = tid; j < CN * BN; j += blockDim.x)
    attn_out[(size_t)b * CN * BN + j] = awf[j];

  // ---- U[c] = sum_n attn[c,n] * band[n,:]  (thread owns d = tid, tid+256) --
  float aR[CN][2] = {}, aI[CN][2] = {};
  for (int n = 0; n < BN; ++n) {
    float r0 = br[(size_t)n * DD + tid];
    float r1 = br[(size_t)n * DD + tid + 256];
    float i0 = bi[(size_t)n * DD + tid];
    float i1 = bi[(size_t)n * DD + tid + 256];
#pragma unroll
    for (int c = 0; c < CN; ++c) {
      float a = aw[c][n];
      aR[c][0] += a * r0;  aR[c][1] += a * r1;
      aI[c][0] += a * i0;  aI[c][1] += a * i1;
    }
  }
#pragma unroll
  for (int c = 0; c < CN; ++c) {
    size_t base = ((size_t)b * CN + c) * K2D;
    ucat[base + tid]            = f2bfu(aR[c][0]);
    ucat[base + tid + 256]      = f2bfu(aR[c][1]);
    ucat[base + DD + tid]       = f2bfu(aI[c][0]);
    ucat[base + DD + tid + 256] = f2bfu(aI[c][1]);
  }
}

// ------ k3: Z = Ucat @ Wcat^T (WMMA, full D per block) + fused distances ----
__global__ void __launch_bounds__(256)
k3_gemm_loss(const unsigned short* __restrict__ ucat,
             const unsigned short* __restrict__ wcatR,
             const unsigned short* __restrict__ wcatI,
             const float* __restrict__ bvr, const float* __restrict__ bvi,
             const float* __restrict__ proto, const int* __restrict__ labels,
             float* __restrict__ lossbuf, float* __restrict__ out)
{
  const int m0   = blockIdx.x * 16;       // rows m = b*8 + c
  const int tid  = threadIdx.x;
  const int wave = tid >> 5;
  const int lane = tid & 31;
  const int half = lane >> 4;
  const int lcol = lane & 15;

  __shared__ uint4 atile4[2048];          // 16 x 1024 bf16 = 32KB
  unsigned short* atile = (unsigned short*)atile4;
  __shared__ float part[2][256][8];       // per-lane partials (deterministic)
  __shared__ float red[2][16];

  // stage A tile to LDS
  {
    const uint4* src = (const uint4*)(ucat + (size_t)m0 * K2D);
    for (int j = tid; j < 2048; j += 256) atile4[j] = src[j];
  }
  __syncthreads();

  const int obase = wave * 64;            // this wave's 64 output cols
  f32x8 accR[4] = {}, accI[4] = {};

  const unsigned short* arow = atile + (size_t)lcol * K2D;
  for (int kb = 0; kb < K2D; kb += 32) {
    bf16x8 lo = *(const bf16x8*)(arow + kb + half * 16);
    bf16x8 hi = *(const bf16x8*)(arow + kb + half * 16 + 8);
    bf16x16 af;
#pragma unroll
    for (int j = 0; j < 8; ++j) { af[j] = lo[j]; af[8 + j] = hi[j]; }
    if (kb + 32 < K2D) {                  // gfx1250 global_prefetch_b8 path
      __builtin_prefetch(wcatR + (size_t)(obase + lcol) * K2D + kb + 32, 0, 1);
      __builtin_prefetch(wcatI + (size_t)(obase + lcol) * K2D + kb + 32, 0, 1);
    }
#pragma unroll
    for (int t = 0; t < 4; ++t) {
      const int o = obase + t * 16 + lcol;
      bf16x16 bR = *(const bf16x16*)(wcatR + (size_t)o * K2D + kb + half * 16);
      bf16x16 bI = *(const bf16x16*)(wcatI + (size_t)o * K2D + kb + half * 16);
      accR[t] = __builtin_amdgcn_wmma_f32_16x16x32_bf16(false, af, false, bR,
                                                        (short)0, accR[t], false, false);
      accI[t] = __builtin_amdgcn_wmma_f32_16x16x32_bf16(false, af, false, bI,
                                                        (short)0, accI[t], false, false);
    }
  }

  // fused epilogue: Z = acc + bias; distance terms; per-lane partial sums
  const float* Pr = proto;
  const float* Pi = proto + (size_t)CN * DD;
  float sD2[8] = {}, sDm[8] = {};
#pragma unroll
  for (int t = 0; t < 4; ++t) {
    const int o = obase + t * 16 + lcol;
    const float bo_r = bvr[o], bo_i = bvi[o];
#pragma unroll
    for (int r = 0; r < 8; ++r) {
      // row M = r + 8*half; m0 % 8 == 0 so class c == r
      const int c = r;
      float zr = accR[t][r] + bo_r;
      float zi = accI[t][r] + bo_i;
      float pr = Pr[(size_t)c * DD + o];
      float pi = Pi[(size_t)c * DD + o];
      float dr = zr - pr, di = zi - pi;
      sD2[r] += dr * dr + di * di;
      float magz = __fsqrt_rn(zr * zr + zi * zi) + 1e-8f;
      float orth = fabsf(zr * pi - zi * pr) / magz;
      float dot  = zr * pr + zi * pi;
      float sgn  = (dot < 0.0f) ? (2.0f * magz - orth) : orth;
      float pmag = __fsqrt_rn(pr * pr + pi * pi);
      sDm[r] += sgn + fabsf(magz - pmag);
    }
  }
#pragma unroll
  for (int r = 0; r < 8; ++r) { part[0][tid][r] = sD2[r]; part[1][tid][r] = sDm[r]; }
  __syncthreads();

  // deterministic fixed-order reduction: 32 threads, one (metric,row) each
  if (tid < 32) {
    const int met = tid >> 4;
    const int M   = tid & 15;
    const int r   = M & 7;
    const int h   = M >> 3;
    float s = 0.0f;
    for (int w = 0; w < 8; ++w)
      for (int lc = 0; lc < 16; ++lc)
        s += part[met][w * 32 + h * 16 + lc][r];
    red[met][M] = s;
  }
  __syncthreads();

  // per-batch finalize: rows [8*tid, 8*tid+8) belong to batch blockIdx.x*2+tid
  if (tid < 2) {
    const int bb = blockIdx.x * 2 + tid;
    float lg[8], d2[8];
#pragma unroll
    for (int c = 0; c < 8; ++c) { d2[c] = red[0][8 * tid + c]; lg[c] = -red[1][8 * tid + c]; }
    float m = -1e30f; int arg = 0;
#pragma unroll
    for (int c = 0; c < 8; ++c) if (lg[c] > m) { m = lg[c]; arg = c; }
    float se = 0.0f;
#pragma unroll
    for (int c = 0; c < 8; ++c) se += __expf(lg[c] - m);
    const float lse = __logf(se);
    const int lab = labels[bb];
    const float loss_cls = -(lg[lab] - m - lse);

    float m2 = -1e30f;
#pragma unroll
    for (int c = 0; c < 8; ++c) m2 = fmaxf(m2, -d2[c]);
    float s2 = 0.0f;
#pragma unroll
    for (int c = 0; c < 8; ++c) s2 += __expf(-d2[c] - m2);
    float sm = 0.0f;
#pragma unroll
    for (int c = 0; c < 8; ++c) sm += (__expf(-d2[c] - m2) / s2) * d2[c];

    lossbuf[bb] = loss_cls + 0.01f * sm;
    out[1 + bb] = (float)arg;             // preds
  }
}

// --------------------- k4: deterministic loss reduction ---------------------
__global__ void __launch_bounds__(256)
k4_reduce(const float* __restrict__ lossbuf, float* __restrict__ out)
{
  __shared__ float p[256];
  float s = 0.0f;
  for (int j = threadIdx.x; j < BB; j += 256) s += lossbuf[j];
  p[threadIdx.x] = s;
  __syncthreads();
  if (threadIdx.x == 0) {
    float t = 0.0f;
    for (int i = 0; i < 256; ++i) t += p[i];
    out[0] = t * (1.0f / (float)BB);
  }
}

// ---------------------------------------------------------------------------
extern "C" void kernel_launch(void* const* d_in, const int* in_sizes, int n_in,
                              void* d_out, int out_size, void* d_ws, size_t ws_size,
                              hipStream_t stream)
{
  (void)in_sizes; (void)n_in; (void)out_size; (void)ws_size;
  const float* band_r = (const float*)d_in[0];
  const float* band_i = (const float*)d_in[1];
  const int*   labels = (const int*)  d_in[2];
  const float* proto  = (const float*)d_in[3];
  const float* Wqr = (const float*)d_in[4];
  const float* Wqi = (const float*)d_in[5];
  const float* bqr = (const float*)d_in[6];
  const float* bqi = (const float*)d_in[7];
  const float* Wkr = (const float*)d_in[8];
  const float* Wki = (const float*)d_in[9];
  const float* bkr = (const float*)d_in[10];
  const float* bki = (const float*)d_in[11];
  const float* Wvr = (const float*)d_in[12];
  const float* Wvi = (const float*)d_in[13];
  const float* bvr = (const float*)d_in[14];
  const float* bvi = (const float*)d_in[15];

  char* ws = (char*)d_ws;
  float*          qr      = (float*)(ws + OFF_QR);
  float*          qi      = (float*)(ws + OFF_QI);
  float*          s0      = (float*)(ws + OFF_S0);
  unsigned short* wscore  = (unsigned short*)(ws + OFF_WSCORE);
  unsigned short* wcatR   = (unsigned short*)(ws + OFF_WCATR);
  unsigned short* wcatI   = (unsigned short*)(ws + OFF_WCATI);
  unsigned short* ucat    = (unsigned short*)(ws + OFF_UCAT);
  float*          lossbuf = (float*)(ws + OFF_LOSS);

  float* out = (float*)d_out;             // [0]=loss, [1..2048]=preds, rest=attn
  float* attn_out = out + 1 + BB;

  k1a_q     <<<CN,   256, 0, stream>>>(proto, Wqr, Wqi, bqr, bqi, qr, qi);
  k1b_fold  <<<16,   256, 0, stream>>>(qr, qi, Wkr, Wki, bkr, bki, wscore, s0);
  k1c_packWv<<<DD,   256, 0, stream>>>(Wvr, Wvi, wcatR, wcatI);
  k2_attn_u <<<BB,   256, 0, stream>>>(band_r, band_i, wscore, s0, attn_out, ucat);
  k3_gemm_loss<<<BB * CN / 16, 256, 0, stream>>>(ucat, wcatR, wcatI, bvr, bvi,
                                                 proto, labels, lossbuf, out);
  k4_reduce <<<1,    256, 0, stream>>>(lossbuf, out);
}
